// TransformerPolicy_6133213299055
// MI455X (gfx1250) — matrix operand order repaired and checked by reading, not compiled, not run on hardware
//
#include <hip/hip_runtime.h>
#include <hip/hip_bf16.h>
#include <math.h>

// ---------------------------------------------------------------------------
// Types for CDNA5 WMMA (wave32): v_wmma_f32_16x16x32_bf16
// ---------------------------------------------------------------------------
typedef __bf16 bf16_t;
typedef __attribute__((ext_vector_type(16))) __bf16 v16bf;
typedef __attribute__((ext_vector_type(8)))  __bf16 v8bf;
typedef __attribute__((ext_vector_type(8)))  float  v8f;
typedef __attribute__((ext_vector_type(4)))  float  v4f;

#define LDK 48   // padded K-stride (bf16 elems) -> 96B rows, 16B-aligned frag loads

__device__ __forceinline__ v16bf cat16(v8bf lo, v8bf hi) {
  union { v16bf v; v8bf h[2]; } u;
  u.h[0] = lo; u.h[1] = hi;
  return u.v;
}
// A-matrix 16x32 bf16 fragment (ISA 7.12.2): lanes 0-15 hold K={0..7,16..23},
// lanes 16-31 hold K={8..15,24..31} of this 32-wide K slice. Row data is
// contiguous in LDS, so each fragment is two aligned 16B loads.
__device__ __forceinline__ v16bf ldfragA(const bf16_t* row, int kb /*0 or 8*/) {
  return cat16(*(const v8bf*)(row + kb), *(const v8bf*)(row + kb + 16));
}
// B-matrix 32x16 bf16 fragment: lane holds column N=lane&15; lanes 0-15 carry
// K={0..7,16..23}, lanes 16-31 carry K={8..15,24..31} (same order as A).
__device__ __forceinline__ v16bf ldfragB(const bf16_t* col, int kb /*0 or 8*/) {
  return cat16(*(const v8bf*)(col + kb), *(const v8bf*)(col + kb + 16));
}

// ---------------------------------------------------------------------------
// Generic GEMM: C[M,N] = epi(A[M,K](lda) @ W[K,N] + bias) (+ resid)
// ACT: 0 none, 1 relu, 2 exact gelu. Block tile 128x128, 8 waves (2M x 4N),
// wave tile 64x32 = 4x2 WMMA tiles, K-step 32. fp32 -> bf16 on LDS stage.
// ---------------------------------------------------------------------------
template <int ACT>
__global__ __launch_bounds__(256) void gemm_bf16_wmma(
    const float* __restrict__ A, long lda,
    const float* __restrict__ W,
    const float* __restrict__ bias,
    const float* __restrict__ resid,
    float* __restrict__ C,
    int M, int N, int K)
{
  __shared__ alignas(32) bf16_t As[128 * LDK];
  __shared__ alignas(32) bf16_t Bs[128 * LDK];

  const int tid   = threadIdx.x;
  const int mBase = blockIdx.y * 128;
  const int nBase = blockIdx.x * 128;
  const int wave  = tid >> 5;
  const int lane  = tid & 31;
  const int waveM = wave & 1;   // 2 waves along M (64 rows each)
  const int waveN = wave >> 1;  // 4 waves along N (32 cols each)
  const int lh    = lane >> 4;
  const int l16   = lane & 15;

  v8f acc[4][2];
#pragma unroll
  for (int mf = 0; mf < 4; ++mf)
#pragma unroll
    for (int nf = 0; nf < 2; ++nf)
      acc[mf][nf] = (v8f){0.f, 0.f, 0.f, 0.f, 0.f, 0.f, 0.f, 0.f};

  const int kTiles = K >> 5;  // K is always a multiple of 32 here
  for (int kt = 0; kt < kTiles; ++kt) {
    const int k0 = kt << 5;

    // Stage A tile (128 x 32) fp32 -> bf16, row-major [r][k]
#pragma unroll
    for (int i = 0; i < 4; ++i) {
      int f  = tid + (i << 8);
      int r  = f >> 3;
      int c4 = (f & 7) << 2;
      int rg = mBase + r;
      v4f a = {0.f, 0.f, 0.f, 0.f};
      if (rg < M) a = *(const v4f*)(A + (long)rg * lda + k0 + c4);
#pragma unroll
      for (int j = 0; j < 4; ++j) As[r * LDK + c4 + j] = (bf16_t)a[j];
    }
    // Stage B tile (32 x 128) fp32 -> bf16, stored transposed [n][k]
#pragma unroll
    for (int i = 0; i < 4; ++i) {
      int f  = tid + (i << 8);
      int kk = f >> 5;
      int n4 = (f & 31) << 2;
      int cg = nBase + n4;
      v4f w = {0.f, 0.f, 0.f, 0.f};
      if (cg + 3 < N) {
        w = *(const v4f*)(W + (long)(k0 + kk) * N + cg);
      } else {
#pragma unroll
        for (int j = 0; j < 4; ++j)
          if (cg + j < N) w[j] = W[(long)(k0 + kk) * N + cg + j];
      }
#pragma unroll
      for (int j = 0; j < 4; ++j) Bs[(n4 + j) * LDK + kk] = (bf16_t)w[j];
    }
    // Prefetch next weight tile rows into cache (global_prefetch_b8)
    if (kt + 1 < kTiles) {
      __builtin_prefetch(W + (long)(k0 + 32 + wave * 4) * N + nBase + lane * 4, 0, 1);
    }
    __syncthreads();

    const int kbA = lh ? 8 : 0;
    const int kbB = lh ? 8 : 0;
    v16bf af[4], bfr[2];
#pragma unroll
    for (int mf = 0; mf < 4; ++mf) {
      int row = waveM * 64 + mf * 16 + l16;
      af[mf] = ldfragA(&As[row * LDK], kbA);
    }
#pragma unroll
    for (int nf = 0; nf < 2; ++nf) {
      int col = waveN * 32 + nf * 16 + l16;
      bfr[nf] = ldfragB(&Bs[col * LDK], kbB);
    }
#pragma unroll
    for (int mf = 0; mf < 4; ++mf)
#pragma unroll
      for (int nf = 0; nf < 2; ++nf)
        acc[mf][nf] = __builtin_amdgcn_wmma_f32_16x16x32_bf16(
            false, af[mf], false, bfr[nf], (short)0, acc[mf][nf], false, false);
    __syncthreads();
  }

  // Epilogue. C/D layout: VGPR j -> row (lh*8 + j), col = lane&15 within tile.
#pragma unroll
  for (int mf = 0; mf < 4; ++mf) {
#pragma unroll
    for (int nf = 0; nf < 2; ++nf) {
      int col = nBase + waveN * 32 + nf * 16 + l16;
      if (col >= N) continue;
      float bia = bias ? bias[col] : 0.f;
#pragma unroll
      for (int j = 0; j < 8; ++j) {
        int row = mBase + waveM * 64 + mf * 16 + lh * 8 + j;
        if (row >= M) continue;
        float x = acc[mf][nf][j] + bia;
        if (ACT == 1) x = fmaxf(x, 0.f);
        else if (ACT == 2) x = 0.5f * x * (1.f + erff(x * 0.70710678118654752f));
        if (resid) x += resid[(long)row * N + col];
        C[(long)row * N + col] = x;
      }
    }
  }
}

// ---------------------------------------------------------------------------
// LayerNorm over H=1024, one 256-thread block per token.
// ---------------------------------------------------------------------------
__global__ __launch_bounds__(256) void ln_kernel(
    const float* __restrict__ x, const float* __restrict__ g,
    const float* __restrict__ b, float* __restrict__ o, int T)
{
  int t = blockIdx.x;
  if (t >= T) return;
  const float* r = x + (long)t * 1024;
  int tid = threadIdx.x;
  float v0[4];
  float s = 0.f, s2 = 0.f;
#pragma unroll
  for (int i = 0; i < 4; ++i) {
    float a = r[tid + i * 256];
    v0[i] = a; s += a; s2 += a * a;
  }
#pragma unroll
  for (int off = 16; off > 0; off >>= 1) {
    s  += __shfl_xor(s, off, 32);
    s2 += __shfl_xor(s2, off, 32);
  }
  __shared__ float ss[8], ss2[8];
  int w = tid >> 5, ln = tid & 31;
  if (ln == 0) { ss[w] = s; ss2[w] = s2; }
  __syncthreads();
  if (tid == 0) {
    float a = 0.f, c = 0.f;
#pragma unroll
    for (int i = 0; i < 8; ++i) { a += ss[i]; c += ss2[i]; }
    float mu = a * (1.f / 1024.f);
    float var = c * (1.f / 1024.f) - mu * mu;
    ss[0] = mu; ss2[0] = rsqrtf(var + 1e-5f);
  }
  __syncthreads();
  float mu = ss[0], rs = ss2[0];
  float* oo = o + (long)t * 1024;
#pragma unroll
  for (int i = 0; i < 4; ++i) {
    int c = tid + i * 256;
    oo[c] = (v0[i] - mu) * rs * g[c] + b[c];
  }
}

// ---------------------------------------------------------------------------
// x[b,s,:] += pos_emb[s,:]   (H=1024, S=32 hardwired powers of two)
// ---------------------------------------------------------------------------
__global__ void addpos_kernel(float* __restrict__ h, const float* __restrict__ pos, int total)
{
  int i = blockIdx.x * 256 + threadIdx.x;
  if (i >= total) return;
  int hc = i & 1023;
  int s  = (i >> 10) & 31;
  h[i] += pos[s * 1024 + hc];
}

// ---------------------------------------------------------------------------
// Attention for one (batch, head): S=32, HD=64. Single wave (EXEC all-ones
// at every WMMA). scores = (q@k^T)*SCALE + rel-bias; softmax; o = attn @ v.
// q/k/v layout: [B, S, NH, HD] fp32.
// ---------------------------------------------------------------------------
__global__ __launch_bounds__(32) void attn_kernel(
    const float* __restrict__ q, const float* __restrict__ k,
    const float* __restrict__ v, const float* __restrict__ rel,
    float* __restrict__ o)
{
  const int S = 32, NH = 16, HD = 64;
  const float SCALE = 0.125f;  // 1/sqrt(64)
  int bh = blockIdx.x;
  int b  = bh >> 4;
  int hh = bh & 15;

  __shared__ alignas(32) bf16_t qb[32 * 72];
  __shared__ alignas(32) bf16_t kb[32 * 72];
  __shared__ alignas(32) bf16_t vb[32 * 72];
  __shared__ float rowq[32 * 64];
  __shared__ float sc[32 * 34];
  __shared__ alignas(32) bf16_t ab[32 * 48];

  int lane = threadIdx.x;
  long base = (((long)b * S) * NH + hh) * HD;  // + s*NH*HD + d

  {  // lane loads sequence row `lane` of q,k,v
    const float* qp = q + base + (long)lane * NH * HD;
    const float* kp = k + base + (long)lane * NH * HD;
    const float* vp = v + base + (long)lane * NH * HD;
    for (int d = 0; d < HD; ++d) {
      float qv = qp[d];
      rowq[lane * 64 + d] = qv;
      qb[lane * 72 + d] = (bf16_t)qv;
      kb[lane * 72 + d] = (bf16_t)kp[d];
      vb[lane * 72 + d] = (bf16_t)vp[d];
    }
  }
  __syncthreads();

  int lh = lane >> 4, l16 = lane & 15;

  // ---- scores = q @ k^T : M=32, N=32, K=64 (two K-steps) ----
  v8f acc[2][2];
#pragma unroll
  for (int mt = 0; mt < 2; ++mt)
#pragma unroll
    for (int nt = 0; nt < 2; ++nt)
      acc[mt][nt] = (v8f){0.f, 0.f, 0.f, 0.f, 0.f, 0.f, 0.f, 0.f};
#pragma unroll
  for (int ks = 0; ks < 64; ks += 32) {
    v16bf af[2], bfr[2];
#pragma unroll
    for (int mt = 0; mt < 2; ++mt) {
      int row = mt * 16 + l16;
      af[mt] = ldfragA(&qb[row * 72 + ks], lh ? 8 : 0);
    }
#pragma unroll
    for (int nt = 0; nt < 2; ++nt) {
      int col = nt * 16 + l16;  // k-sequence index; elements run over d
      bfr[nt] = ldfragB(&kb[col * 72 + ks], lh ? 8 : 0);
    }
#pragma unroll
    for (int mt = 0; mt < 2; ++mt)
#pragma unroll
      for (int nt = 0; nt < 2; ++nt)
        acc[mt][nt] = __builtin_amdgcn_wmma_f32_16x16x32_bf16(
            false, af[mt], false, bfr[nt], (short)0, acc[mt][nt], false, false);
  }
#pragma unroll
  for (int mt = 0; mt < 2; ++mt)
#pragma unroll
    for (int nt = 0; nt < 2; ++nt)
#pragma unroll
      for (int j = 0; j < 8; ++j)
        sc[(mt * 16 + lh * 8 + j) * 34 + nt * 16 + l16] = acc[mt][nt][j] * SCALE;
  __syncthreads();

  // ---- relative-position bias + softmax: lane owns row r ----
  {
    int r = lane;
    float* srow = &sc[r * 34];
    for (int kk = 0; kk < 32; ++kk) {
      int di = r - kk;
      di = di < -32 ? -32 : (di > 32 ? 32 : di);
      const float* rv = rel + (di + 32) * 64;
      float a = 0.f;
#pragma unroll
      for (int d = 0; d < 64; ++d) a += rowq[r * 64 + d] * rv[d];
      srow[kk] += a;
    }
    float mx = srow[0];
    for (int kk = 1; kk < 32; ++kk) mx = fmaxf(mx, srow[kk]);
    float sm = 0.f;
    for (int kk = 0; kk < 32; ++kk) { float e = __expf(srow[kk] - mx); srow[kk] = e; sm += e; }
    float inv = 1.f / sm;
    for (int kk = 0; kk < 32; ++kk) ab[r * 48 + kk] = (bf16_t)(srow[kk] * inv);
  }
  __syncthreads();

  // ---- o = attn @ v : M=32, N=64, K=32 ----
  v8f oacc[2][4];
#pragma unroll
  for (int mt = 0; mt < 2; ++mt)
#pragma unroll
    for (int nt = 0; nt < 4; ++nt)
      oacc[mt][nt] = (v8f){0.f, 0.f, 0.f, 0.f, 0.f, 0.f, 0.f, 0.f};
  v16bf aaf[2];
#pragma unroll
  for (int mt = 0; mt < 2; ++mt)
    aaf[mt] = ldfragA(&ab[(mt * 16 + l16) * 48], lh ? 8 : 0);
#pragma unroll
  for (int nt = 0; nt < 4; ++nt) {
    int col = nt * 16 + l16;  // d dimension
    int kbB = lh ? 8 : 0;     // s-index base
    v16bf bv;
#pragma unroll
    for (int e = 0; e < 16; ++e) bv[e] = vb[(kbB + e + (e & 8)) * 72 + col];
#pragma unroll
    for (int mt = 0; mt < 2; ++mt)
      oacc[mt][nt] = __builtin_amdgcn_wmma_f32_16x16x32_bf16(
          false, aaf[mt], false, bv, (short)0, oacc[mt][nt], false, false);
  }
#pragma unroll
  for (int mt = 0; mt < 2; ++mt)
#pragma unroll
    for (int nt = 0; nt < 4; ++nt)
#pragma unroll
      for (int j = 0; j < 8; ++j) {
        int s_ = mt * 16 + lh * 8 + j;
        int d  = nt * 16 + l16;
        o[base + (long)s_ * NH * HD + d] = oacc[mt][nt][j];
      }
}

// ---------------------------------------------------------------------------
// Host-side orchestration
// ---------------------------------------------------------------------------
static inline void launch_gemm(const float* A, long lda, const float* W,
                               const float* bias, const float* resid, float* C,
                               int M, int N, int K, int act, hipStream_t s)
{
  dim3 g((N + 127) / 128, (M + 127) / 128), b(256);
  if (act == 0)      gemm_bf16_wmma<0><<<g, b, 0, s>>>(A, lda, W, bias, resid, C, M, N, K);
  else if (act == 1) gemm_bf16_wmma<1><<<g, b, 0, s>>>(A, lda, W, bias, resid, C, M, N, K);
  else               gemm_bf16_wmma<2><<<g, b, 0, s>>>(A, lda, W, bias, resid, C, M, N, K);
}

extern "C" void kernel_launch(void* const* d_in, const int* in_sizes, int n_in,
                              void* d_out, int out_size, void* d_ws, size_t ws_size,
                              hipStream_t stream)
{
  (void)in_sizes; (void)n_in; (void)out_size; (void)ws_size;
  const int B = 512, S = 32, H = 1024, NH = 16, HD = 64, F = 2048, L = 6;
  const int TOK = B * S;  // 16384

  const float* obs    = (const float*)d_in[0];
  const float* ref    = (const float*)d_in[1];
  const float* obs_w0 = (const float*)d_in[2];
  const float* obs_b0 = (const float*)d_in[3];
  const float* obs_w1 = (const float*)d_in[4];
  const float* obs_b1 = (const float*)d_in[5];
  const float* obs_w2 = (const float*)d_in[6];
  const float* obs_b2 = (const float*)d_in[7];
  const float* ref_w0 = (const float*)d_in[8];
  const float* ref_b0 = (const float*)d_in[9];
  const float* ref_w1 = (const float*)d_in[10];
  const float* ref_b1 = (const float*)d_in[11];
  const float* ref_w2 = (const float*)d_in[12];
  const float* ref_b2 = (const float*)d_in[13];
  const float* pos    = (const float*)d_in[14];
  const float* Wq     = (const float*)d_in[15];
  const float* Wk     = (const float*)d_in[16];
  const float* Wv     = (const float*)d_in[17];
  const float* Wo     = (const float*)d_in[18];
  const float* bo     = (const float*)d_in[19];
  const float* rel    = (const float*)d_in[20];
  const float* ff_w1  = (const float*)d_in[21];
  const float* ff_b1  = (const float*)d_in[22];
  const float* ff_w2  = (const float*)d_in[23];
  const float* ff_b2  = (const float*)d_in[24];
  const float* ln1g   = (const float*)d_in[25];
  const float* ln1b   = (const float*)d_in[26];
  const float* ln2g   = (const float*)d_in[27];
  const float* ln2b   = (const float*)d_in[28];
  const float* act_w0 = (const float*)d_in[29];
  const float* act_b0 = (const float*)d_in[30];
  const float* act_w1 = (const float*)d_in[31];
  const float* act_b1 = (const float*)d_in[32];
  const float* act_w2 = (const float*)d_in[33];
  const float* act_b2 = (const float*)d_in[34];

  float* ws = (float*)d_ws;
  const size_t SZ = (size_t)TOK * H;  // 16M floats
  float* h   = ws;             // [TOK, H]
  float* xn  = ws + 1 * SZ;    // [TOK, H]
  float* q   = ws + 2 * SZ;    // [TOK, H] == [B,S,NH,HD]
  float* k   = ws + 3 * SZ;
  float* v   = ws + 4 * SZ;
  float* tmp = ws + 5 * SZ;    // 2*SZ floats scratch (FFN hidden / MLP / attn-out)
  float* t0  = tmp;                         // [TOK, 512]
  float* t1  = tmp + (size_t)TOK * 512;     // [TOK, 256]

  // ---- input encoders ----
  launch_gemm(obs, 256, obs_w0, obs_b0, nullptr, t0, TOK, 512, 256, 1, stream);
  launch_gemm(t0, 512, obs_w1, obs_b1, nullptr, t1, TOK, 256, 512, 1, stream);
  launch_gemm(t1, 256, obs_w2, obs_b2, nullptr, h,  TOK, H,   256, 0, stream);
  launch_gemm(ref, 128, ref_w0, ref_b0, nullptr, t0, TOK, 256, 128, 1, stream);
  launch_gemm(t0, 256, ref_w1, ref_b1, nullptr, t1, TOK, 256, 256, 1, stream);
  launch_gemm(t1, 256, ref_w2, ref_b2, h, h,       TOK, H,   256, 0, stream);
  addpos_kernel<<<(TOK * H + 255) / 256, 256, 0, stream>>>(h, pos, TOK * H);

  // ---- transformer blocks ----
  for (int l = 0; l < L; ++l) {
    const float* wq = Wq + (size_t)l * H * H;
    const float* wk = Wk + (size_t)l * H * H;
    const float* wv = Wv + (size_t)l * H * H;
    const float* wo = Wo + (size_t)l * H * H;
    ln_kernel<<<TOK, 256, 0, stream>>>(h, ln1g + l * H, ln1b + l * H, xn, TOK);
    launch_gemm(xn, H, wq, nullptr, nullptr, q, TOK, H, H, 0, stream);
    launch_gemm(xn, H, wk, nullptr, nullptr, k, TOK, H, H, 0, stream);
    launch_gemm(xn, H, wv, nullptr, nullptr, v, TOK, H, H, 0, stream);
    attn_kernel<<<B * NH, 32, 0, stream>>>(q, k, v, rel + (size_t)l * 65 * 64, tmp);
    launch_gemm(tmp, H, wo, bo + l * H, h, h, TOK, H, H, 0, stream);  // h += o@Wo + bo
    ln_kernel<<<TOK, 256, 0, stream>>>(h, ln2g + l * H, ln2b + l * H, xn, TOK);
    launch_gemm(xn, H, ff_w1 + (size_t)l * H * F, ff_b1 + l * F, nullptr, tmp,
                TOK, F, H, 2, stream);                                 // gelu
    launch_gemm(tmp, F, ff_w2 + (size_t)l * F * H, ff_b2 + l * H, h, h,
                TOK, H, F, 0, stream);                                 // h += ff
  }

  // ---- action head on last token: A = h[:, S-1, :], row stride S*H ----
  float* a0 = tmp;                     // [B, 256]
  float* a1 = tmp + (size_t)B * 256;   // [B, 128]
  launch_gemm(h + (size_t)(S - 1) * H, (long)S * H, act_w0, act_b0, nullptr, a0,
              B, 256, H, 1, stream);
  launch_gemm(a0, 256, act_w1, act_b1, nullptr, a1, B, 128, 256, 1, stream);
  launch_gemm(a1, 128, act_w2, act_b2, nullptr, (float*)d_out, B, 32, 128, 0, stream);
}
